// TransformerBlock_85341000171856
// MI455X (gfx1250) — matrix operand order repaired and checked by reading, not compiled, not run on hardware
//
#include <hip/hip_runtime.h>
#include <hip/hip_bf16.h>
#include <stdint.h>

// ---------------------------------------------------------------------------
// Problem constants (DistilBERT-style block)
// ---------------------------------------------------------------------------
#define BS_    2
#define SEQ_   2048
#define DIM_   1024
#define HID_   4096
#define NH_    16
#define HD_    64          // head dim
#define TOK_   (BS_ * SEQ_)

typedef __bf16 bf16_t;
typedef bf16_t v16bf __attribute__((ext_vector_type(16)));
typedef float  v8f   __attribute__((ext_vector_type(8)));

union Frag16 {             // one 16x32 A-fragment or 32x16 B-fragment (bf16)
    uint4 q[2];
    v16bf v;
};

__device__ __forceinline__ uint16_t f32_bf16(float f) {
    uint32_t u = __float_as_uint(f);
    uint32_t r = u + 0x7FFFu + ((u >> 16) & 1u);   // round-to-nearest-even
    return (uint16_t)(r >> 16);
}

// ---------------------------------------------------------------------------
// CDNA5 async global->LDS (ASYNCcnt-tracked, no VGPR staging).
// VGLOBAL encoding: VDST = LDS byte address VGPR, VADDR = 64-bit global addr.
// Generic pointers to __shared__ carry the LDS byte offset in their low 32
// bits (flat-LDS aperture mapping), so truncation yields the VDST value.
// ---------------------------------------------------------------------------
__device__ __forceinline__ void async_ld_b128(const void* lds_dst, const void* gsrc) {
    asm volatile("global_load_async_to_lds_b128 %0, %1, off"
                 :: "v"((uint32_t)(uintptr_t)lds_dst), "v"(gsrc)
                 : "memory");
}
__device__ __forceinline__ void wait_async0() {
    asm volatile("s_wait_asynccnt 0x0" ::: "memory");
}

// ---------------------------------------------------------------------------
// fp32 -> bf16 conversion
// ---------------------------------------------------------------------------
__global__ __launch_bounds__(256) void f32_to_bf16_k(const float* __restrict__ s,
                                                     uint16_t* __restrict__ d, int n) {
    int i = blockIdx.x * 256 + threadIdx.x;
    if (i < n) d[i] = f32_bf16(s[i]);
}

// ---------------------------------------------------------------------------
// bf16 GEMM: C[M,N] = A[M,K] @ B[K,N] + bias, epilogue variants.
//   EPI 0: fp32 out (bias)
//   EPI 1: bf16 out ((acc+bias)*scale)     -> Q/K/V projections
//   EPI 2: bf16 out (gelu(acc+bias))       -> FFN1
// Block tile 128x128, K-step 32, 256 threads = 8 waves, wave tile 32x64.
// A tiles: async global->LDS, double-buffered (copy of tile t+1 overlaps
// WMMA on tile t). B tiles: register transpose into [n][k] LDS layout.
// ---------------------------------------------------------------------------
#define BM 128
#define BN 128
#define BK 32
#define LDSA 40    // padded LDS strides (multiple of 8 -> 16B-aligned frag loads)
#define LDSB 40

template<int EPI>
__global__ __launch_bounds__(256) void gemm_bf16_k(
    const uint16_t* __restrict__ A, const uint16_t* __restrict__ B,
    const float* __restrict__ bias, float scale,
    float* __restrict__ Cf, uint16_t* __restrict__ Ch,
    int M, int N, int K)
{
    __shared__ uint16_t lsA[2][BM * LDSA];     // [m][k], double-buffered
    __shared__ uint16_t lsB[2][BN * LDSB];     // transposed [n][k], double-buffered

    const int tid  = threadIdx.x;
    const int lane = tid & 31;
    const int wave = tid >> 5;
    const int mBase = blockIdx.y * BM;
    const int nBase = blockIdx.x * BN;
    const int wm = wave >> 1;                  // 0..3 : 32-row slice
    const int wn = wave & 1;                   // 0..1 : 64-col slice

    const int n16  = lane & 15;
    const int hsel = lane >> 4;                // wave half
    const int a_kb = hsel * 16;                // A-frag K base  (contiguous K order)
    const int b_kb = hsel * 16;                // B-frag K base  (contiguous K order)

    // staging assignments
    const int ar = tid >> 1, ac = (tid & 1) * 16;     // A: 2 thr/row, 16 elems
    const int br = tid >> 3, bc = (tid & 7) * 16;     // B: 8 thr/row, 16 elems

    v8f acc[2][4];
#pragma unroll
    for (int i = 0; i < 2; i++)
#pragma unroll
        for (int j = 0; j < 4; j++) acc[i][j] = (v8f){0,0,0,0,0,0,0,0};

    auto stage = [&](int k0, int buf) {
        // A tile: async DMA straight into LDS (two b128 per thread)
        const uint16_t* gA = A + (size_t)(mBase + ar) * K + k0 + ac;
        async_ld_b128(&lsA[buf][ar * LDSA + ac],     gA);
        async_ld_b128(&lsA[buf][ar * LDSA + ac + 8], gA + 8);
        // B tile transposed [n][k]: register scatter
        uint16_t tmp[16];
        const uint4* gB = (const uint4*)(B + (size_t)(k0 + br) * N + nBase + bc);
        *(uint4*)&tmp[0] = gB[0];
        *(uint4*)&tmp[8] = gB[1];
#pragma unroll
        for (int e = 0; e < 16; e++) lsB[buf][(bc + e) * LDSB + br] = tmp[e];
        // L2 prefetch of the tile after next (global_prefetch_b8)
        if (k0 + BK < K)
            __builtin_prefetch(B + (size_t)(k0 + BK + br) * N + nBase + bc, 0, 0);
    };

    const int nT = K / BK;
    stage(0, 0);

    for (int t = 0; t < nT; ++t) {
        const int buf = t & 1;
        wait_async0();         // this wave's async loads for tile t are in LDS
        __syncthreads();       // every wave's tile-t staging (async + DS) done
        if (t + 1 < nT) stage((t + 1) * BK, buf ^ 1);   // overlap with WMMA

        // ---- fragments from LDS (contiguous K order per lane half) ----
        Frag16 af[2], bf[4];
#pragma unroll
        for (int mf = 0; mf < 2; mf++) {
            int row = wm * 32 + mf * 16 + n16;
            af[mf].q[0] = *(const uint4*)&lsA[buf][row * LDSA + a_kb];      // K=kb..kb+7
            af[mf].q[1] = *(const uint4*)&lsA[buf][row * LDSA + a_kb + 8];  // K=kb+8..kb+15
        }
#pragma unroll
        for (int nf = 0; nf < 4; nf++) {
            int col = wn * 64 + nf * 16 + n16;
            bf[nf].q[0] = *(const uint4*)&lsB[buf][col * LDSB + b_kb];      // K=kb..kb+7
            bf[nf].q[1] = *(const uint4*)&lsB[buf][col * LDSB + b_kb + 8];  // K=kb+8..kb+15
        }
#pragma unroll
        for (int mf = 0; mf < 2; mf++)
#pragma unroll
            for (int nf = 0; nf < 4; nf++)
                acc[mf][nf] = __builtin_amdgcn_wmma_f32_16x16x32_bf16(
                    false, af[mf].v, false, bf[nf].v, (short)0, acc[mf][nf],
                    false, false);
        // no second barrier: tile-t reads are consumed by the WMMAs above,
        // and buffer t+2 staging only begins after the next top-of-loop barrier
    }

    // ---- epilogue ----
    const int half8 = hsel * 8;
#pragma unroll
    for (int mf = 0; mf < 2; mf++) {
#pragma unroll
        for (int nf = 0; nf < 4; nf++) {
            int col = nBase + wn * 64 + nf * 16 + n16;
            float bv = bias[col];
#pragma unroll
            for (int v = 0; v < 8; v++) {
                int row = mBase + wm * 32 + mf * 16 + half8 + v;
                float val = acc[mf][nf][v] + bv;
                if (EPI == 0) {
                    Cf[(size_t)row * N + col] = val;
                } else if (EPI == 1) {
                    Ch[(size_t)row * N + col] = f32_bf16(val * scale);
                } else {
                    float g = 0.5f * val * (1.0f + erff(val * 0.70710678118654752f));
                    Ch[(size_t)row * N + col] = f32_bf16(g);
                }
            }
        }
    }
}

// ---------------------------------------------------------------------------
// Flash attention: per (b, h, 128-row Q block). 256 threads = 8 waves,
// each wave owns 16 Q rows. 64-key K/V tiles streamed through LDS.
// Q and K tiles arrive via async global->LDS DMA; V is transposed on the fly.
// Q is pre-scaled by 1/sqrt(HD) in its projection GEMM.
// ---------------------------------------------------------------------------
#define AQ 128       // Q rows per block
#define AK 64        // keys per tile
#define LQS 72       // padded LDS stride (multiple of 8)

__global__ __launch_bounds__(256) void attn_k(
    const uint16_t* __restrict__ Q, const uint16_t* __restrict__ Kmat,
    const uint16_t* __restrict__ V, const int* __restrict__ mask,
    uint16_t* __restrict__ ctx, int seq)
{
    __shared__ uint16_t lsQ[AQ * LQS];          // [q][d]
    __shared__ uint16_t lsK[AK * LQS];          // [key][d]
    __shared__ uint16_t lsVt[HD_ * LQS];        // transposed: [d][key]
    __shared__ uint16_t lsP[8 * 16 * LQS];      // per-wave P staging [row][key]
    __shared__ float    lsBias[AK];

    const int tid  = threadIdx.x;
    const int lane = tid & 31;
    const int wave = tid >> 5;
    const int b = blockIdx.z, h = blockIdx.y;
    const int qBase = blockIdx.x * AQ;
    const int n16  = lane & 15;
    const int hsel = lane >> 4;
    const int a_kb = hsel * 16;
    const int b_kb = hsel * 16;

    const size_t headOff = (size_t)h * HD_;
    const size_t bOff    = (size_t)b * seq * DIM_;

    // ---- Q block (128 x 64 bf16) via async DMA ----
    {
        int r = tid >> 1, c = (tid & 1) * 32;
        const uint16_t* g = Q + bOff + (size_t)(qBase + r) * DIM_ + headOff + c;
        async_ld_b128(&lsQ[r * LQS + c],      g);
        async_ld_b128(&lsQ[r * LQS + c + 8],  g + 8);
        async_ld_b128(&lsQ[r * LQS + c + 16], g + 16);
        async_ld_b128(&lsQ[r * LQS + c + 24], g + 24);
    }
    wait_async0();
    __syncthreads();

    // per-wave Q A-fragments, d0 = 0 and 32 (K-dim = head dim)
    Frag16 qf[2];
    {
        int row = wave * 16 + n16;
#pragma unroll
        for (int df = 0; df < 2; df++) {
            qf[df].q[0] = *(const uint4*)&lsQ[row * LQS + df * 32 + a_kb];
            qf[df].q[1] = *(const uint4*)&lsQ[row * LQS + df * 32 + a_kb + 8];
        }
    }

    float mrow[8], lrow[8];
    v8f oacc[4];
#pragma unroll
    for (int v = 0; v < 8; v++) { mrow[v] = -3.0e38f; lrow[v] = 0.0f; }
#pragma unroll
    for (int df = 0; df < 4; df++) oacc[df] = (v8f){0,0,0,0,0,0,0,0};

    for (int kt = 0; kt < seq; kt += AK) {
        // ---- stage K tile [key][d] via async DMA ----
        {
            int r = tid >> 2, c = (tid & 3) * 16;
            const uint16_t* g = Kmat + bOff + (size_t)(kt + r) * DIM_ + headOff + c;
            async_ld_b128(&lsK[r * LQS + c],     g);
            async_ld_b128(&lsK[r * LQS + c + 8], g + 8);
        }
        // ---- stage V tile transposed [d][key] ----
        {
            int r = tid >> 2, c = (tid & 3) * 16;
            uint16_t tmp[16];
            const uint4* g = (const uint4*)(V + bOff + (size_t)(kt + r) * DIM_ + headOff + c);
            *(uint4*)&tmp[0] = g[0];
            *(uint4*)&tmp[8] = g[1];
#pragma unroll
            for (int e = 0; e < 16; e++) lsVt[(c + e) * LQS + r] = tmp[e];
        }
        if (tid < AK) lsBias[tid] = (mask[b * seq + kt + tid] != 0) ? 0.0f : -1.0e30f;
        wait_async0();
        __syncthreads();

        // ---- S = Q K^T (+mask bias), 4 col-fragments of 16 keys each ----
        v8f sfr[4];
#pragma unroll
        for (int nf = 0; nf < 4; nf++) {
            int key = nf * 16 + n16;
            Frag16 kf0, kf1;
            kf0.q[0] = *(const uint4*)&lsK[key * LQS + b_kb];
            kf0.q[1] = *(const uint4*)&lsK[key * LQS + b_kb + 8];
            kf1.q[0] = *(const uint4*)&lsK[key * LQS + 32 + b_kb];
            kf1.q[1] = *(const uint4*)&lsK[key * LQS + 32 + b_kb + 8];
            v8f s = (v8f){0,0,0,0,0,0,0,0};
            s = __builtin_amdgcn_wmma_f32_16x16x32_bf16(false, qf[0].v, false, kf0.v,
                                                        (short)0, s, false, false);
            s = __builtin_amdgcn_wmma_f32_16x16x32_bf16(false, qf[1].v, false, kf1.v,
                                                        (short)0, s, false, false);
            float kb = lsBias[nf * 16 + n16];
#pragma unroll
            for (int v = 0; v < 8; v++) s[v] += kb;
            sfr[nf] = s;
        }

        // ---- online softmax: row stats live replicated in each 16-lane half ----
        float rm[8];
#pragma unroll
        for (int v = 0; v < 8; v++) {
            float m = sfr[0][v];
#pragma unroll
            for (int nf = 1; nf < 4; nf++) m = fmaxf(m, sfr[nf][v]);
#pragma unroll
            for (int off = 1; off < 16; off <<= 1)
                m = fmaxf(m, __shfl_xor(m, off, 32));
            rm[v] = m;
        }
        float alpha[8], rs[8];
#pragma unroll
        for (int v = 0; v < 8; v++) {
            float mn = fmaxf(mrow[v], rm[v]);
            alpha[v] = __expf(mrow[v] - mn);
            mrow[v]  = mn;
            rs[v]    = 0.0f;
        }
        // P = exp(S - m), staged to LDS in [row][key] for A-fragment reload
        const int pbase = wave * 16 * LQS;
#pragma unroll
        for (int nf = 0; nf < 4; nf++) {
#pragma unroll
            for (int v = 0; v < 8; v++) {
                float p = __expf(sfr[nf][v] - mrow[v]);
                rs[v] += p;
                lsP[pbase + (hsel * 8 + v) * LQS + nf * 16 + n16] = f32_bf16(p);
            }
        }
#pragma unroll
        for (int v = 0; v < 8; v++) {
            float s = rs[v];
#pragma unroll
            for (int off = 1; off < 16; off <<= 1) s += __shfl_xor(s, off, 32);
            lrow[v] = lrow[v] * alpha[v] + s;
        }
#pragma unroll
        for (int df = 0; df < 4; df++)
#pragma unroll
            for (int v = 0; v < 8; v++) oacc[df][v] *= alpha[v];

        // wait for this wave's P stores before reloading (per-wave LDS region)
        asm volatile("s_wait_dscnt 0" ::: "memory");

        Frag16 pf[2];
#pragma unroll
        for (int kf = 0; kf < 2; kf++) {
            pf[kf].q[0] = *(const uint4*)&lsP[pbase + n16 * LQS + kf * 32 + a_kb];
            pf[kf].q[1] = *(const uint4*)&lsP[pbase + n16 * LQS + kf * 32 + a_kb + 8];
        }
        // ---- O += P @ V ----
#pragma unroll
        for (int df = 0; df < 4; df++) {
            int d = df * 16 + n16;
            Frag16 vf0, vf1;
            vf0.q[0] = *(const uint4*)&lsVt[d * LQS + b_kb];
            vf0.q[1] = *(const uint4*)&lsVt[d * LQS + b_kb + 8];
            vf1.q[0] = *(const uint4*)&lsVt[d * LQS + 32 + b_kb];
            vf1.q[1] = *(const uint4*)&lsVt[d * LQS + 32 + b_kb + 8];
            oacc[df] = __builtin_amdgcn_wmma_f32_16x16x32_bf16(false, pf[0].v, false, vf0.v,
                                                               (short)0, oacc[df], false, false);
            oacc[df] = __builtin_amdgcn_wmma_f32_16x16x32_bf16(false, pf[1].v, false, vf1.v,
                                                               (short)0, oacc[df], false, false);
        }
        __syncthreads();
    }

    // ---- normalize and store ctx (bf16, [b][s][h*64+d] layout) ----
#pragma unroll
    for (int df = 0; df < 4; df++) {
#pragma unroll
        for (int v = 0; v < 8; v++) {
            int row = qBase + wave * 16 + hsel * 8 + v;
            int col = (int)headOff + df * 16 + n16;
            float o = oacc[df][v] / lrow[v];
            ctx[bOff + (size_t)row * DIM_ + col] = f32_bf16(o);
        }
    }
}

// ---------------------------------------------------------------------------
// Fused residual add + LayerNorm. One block per token row (dim=1024).
// Emits fp32 (for later residual) and optional bf16 (for next GEMM).
// ---------------------------------------------------------------------------
__device__ __forceinline__ float block_sum(float v, float* red) {
    int lane = threadIdx.x & 31, wave = threadIdx.x >> 5;
#pragma unroll
    for (int off = 16; off > 0; off >>= 1) v += __shfl_xor(v, off, 32);
    if (lane == 0) red[wave] = v;
    __syncthreads();
    float s = 0.0f;
#pragma unroll
    for (int i = 0; i < 8; i++) s += red[i];
    __syncthreads();
    return s;
}

__global__ __launch_bounds__(256) void add_ln_k(
    const float* __restrict__ a, const float* __restrict__ r,
    const float* __restrict__ gamma, const float* __restrict__ beta,
    float* __restrict__ out32, uint16_t* __restrict__ out16)
{
    __shared__ float red[8];
    const int tid = threadIdx.x;
    const size_t base = (size_t)blockIdx.x * DIM_;
    float vals[4];
    float s = 0.0f;
#pragma unroll
    for (int i = 0; i < 4; i++) {
        int c = tid + i * 256;
        vals[i] = a[base + c] + r[base + c];
        s += vals[i];
    }
    s = block_sum(s, red);
    float mu = s * (1.0f / DIM_);
    float vs = 0.0f;
#pragma unroll
    for (int i = 0; i < 4; i++) { float d = vals[i] - mu; vs += d * d; }
    vs = block_sum(vs, red);
    float inv = rsqrtf(vs * (1.0f / DIM_) + 1e-12f);
#pragma unroll
    for (int i = 0; i < 4; i++) {
        int c = tid + i * 256;
        float y = (vals[i] - mu) * inv * gamma[c] + beta[c];
        out32[base + c] = y;
        if (out16) out16[base + c] = f32_bf16(y);
    }
}

// ---------------------------------------------------------------------------
// Host orchestration
// ---------------------------------------------------------------------------
extern "C" void kernel_launch(void* const* d_in, const int* in_sizes, int n_in,
                              void* d_out, int out_size, void* d_ws, size_t ws_size,
                              hipStream_t stream) {
    (void)in_sizes; (void)n_in; (void)out_size; (void)ws_size;

    const float* x    = (const float*)d_in[0];
    const int*   am   = (const int*)  d_in[1];
    const float* Wq   = (const float*)d_in[2];
    const float* bq   = (const float*)d_in[3];
    const float* Wk   = (const float*)d_in[4];
    const float* bk   = (const float*)d_in[5];
    const float* Wv   = (const float*)d_in[6];
    const float* bv   = (const float*)d_in[7];
    const float* Wo   = (const float*)d_in[8];
    const float* bo   = (const float*)d_in[9];
    const float* ln1g = (const float*)d_in[10];
    const float* ln1b = (const float*)d_in[11];
    const float* W1   = (const float*)d_in[12];
    const float* b1   = (const float*)d_in[13];
    const float* W2   = (const float*)d_in[14];
    const float* b2   = (const float*)d_in[15];
    const float* ln2g = (const float*)d_in[16];
    const float* ln2b = (const float*)d_in[17];
    float* out = (float*)d_out;

    // workspace carve-out (~152 MB total)
    char* w = (char*)d_ws;
    auto carve = [&](size_t bytes) -> char* {
        char* p = w;
        w += (bytes + 255) & ~(size_t)255;
        return p;
    };
    uint16_t* xb   = (uint16_t*)carve((size_t)TOK_ * DIM_ * 2);
    uint16_t* Wqb  = (uint16_t*)carve((size_t)DIM_ * DIM_ * 2);
    uint16_t* Wkb  = (uint16_t*)carve((size_t)DIM_ * DIM_ * 2);
    uint16_t* Wvb  = (uint16_t*)carve((size_t)DIM_ * DIM_ * 2);
    uint16_t* Wob  = (uint16_t*)carve((size_t)DIM_ * DIM_ * 2);
    uint16_t* W1b  = (uint16_t*)carve((size_t)DIM_ * HID_ * 2);
    uint16_t* W2b  = (uint16_t*)carve((size_t)HID_ * DIM_ * 2);
    uint16_t* Qb   = (uint16_t*)carve((size_t)TOK_ * DIM_ * 2);
    uint16_t* Kb   = (uint16_t*)carve((size_t)TOK_ * DIM_ * 2);
    uint16_t* Vb   = (uint16_t*)carve((size_t)TOK_ * DIM_ * 2);
    uint16_t* ctxb = (uint16_t*)carve((size_t)TOK_ * DIM_ * 2);
    float*    sa   = (float*)   carve((size_t)TOK_ * DIM_ * 4);
    float*    h1   = (float*)   carve((size_t)TOK_ * DIM_ * 4);
    uint16_t* h1b  = (uint16_t*)carve((size_t)TOK_ * DIM_ * 2);
    uint16_t* gb   = (uint16_t*)carve((size_t)TOK_ * HID_ * 2);
    float*    f2   = (float*)   carve((size_t)TOK_ * DIM_ * 4);

    auto cvt = [&](const float* s, uint16_t* d, int n) {
        f32_to_bf16_k<<<(n + 255) / 256, 256, 0, stream>>>(s, d, n);
    };
    cvt(x,  xb,  TOK_ * DIM_);
    cvt(Wq, Wqb, DIM_ * DIM_);
    cvt(Wk, Wkb, DIM_ * DIM_);
    cvt(Wv, Wvb, DIM_ * DIM_);
    cvt(Wo, Wob, DIM_ * DIM_);
    cvt(W1, W1b, DIM_ * HID_);
    cvt(W2, W2b, HID_ * DIM_);

    const dim3 blk(256);
    // QKV projections (Q folds in 1/sqrt(64))
    gemm_bf16_k<1><<<dim3(DIM_ / BN, TOK_ / BM), blk, 0, stream>>>(
        xb, Wqb, bq, 0.125f, nullptr, Qb, TOK_, DIM_, DIM_);
    gemm_bf16_k<1><<<dim3(DIM_ / BN, TOK_ / BM), blk, 0, stream>>>(
        xb, Wkb, bk, 1.0f, nullptr, Kb, TOK_, DIM_, DIM_);
    gemm_bf16_k<1><<<dim3(DIM_ / BN, TOK_ / BM), blk, 0, stream>>>(
        xb, Wvb, bv, 1.0f, nullptr, Vb, TOK_, DIM_, DIM_);

    // flash attention
    attn_k<<<dim3(SEQ_ / AQ, NH_, BS_), blk, 0, stream>>>(Qb, Kb, Vb, am, ctxb, SEQ_);

    // output projection (fp32 out)
    gemm_bf16_k<0><<<dim3(DIM_ / BN, TOK_ / BM), blk, 0, stream>>>(
        ctxb, Wob, bo, 1.0f, sa, nullptr, TOK_, DIM_, DIM_);

    // residual + LN1 -> h1 (fp32) + h1b (bf16)
    add_ln_k<<<TOK_, blk, 0, stream>>>(sa, x, ln1g, ln1b, h1, h1b);

    // FFN1 with fused erf-GELU -> bf16
    gemm_bf16_k<2><<<dim3(HID_ / BN, TOK_ / BM), blk, 0, stream>>>(
        h1b, W1b, b1, 1.0f, nullptr, gb, TOK_, HID_, DIM_);
    // FFN2 -> fp32
    gemm_bf16_k<0><<<dim3(DIM_ / BN, TOK_ / BM), blk, 0, stream>>>(
        gb, W2b, b2, 1.0f, f2, nullptr, TOK_, DIM_, HID_);

    // residual + LN2 -> final output
    add_ln_k<<<TOK_, blk, 0, stream>>>(f2, h1, ln2g, ln2b, out, nullptr);
}
